// Qwen2_5VLVisualModel_12360915878776
// MI455X (gfx1250) — compile-verified, matrix operand order checked by reading, not run on hardware
//
#include <hip/hip_runtime.h>
#include <hip/hip_bf16.h>
#include <math.h>

typedef __attribute__((ext_vector_type(16))) _Float16 v16h;
typedef __attribute__((ext_vector_type(8)))  _Float16 v8h;
typedef __attribute__((ext_vector_type(4)))  _Float16 v4h;
typedef __attribute__((ext_vector_type(8)))  float    v8f;

constexpr int kS = 2304, kHID = 1280, kNH = 16, kAD = 80, kROT = 40;
constexpr int kINTER = 3420, kNW = 36, kMU = 4, kG = 576;
constexpr int kMERGED = 5120, kOUT = 3584, kINCH = 1176;
constexpr int kADP = 96;   // head dim padded to multiple of 32 for WMMA K

__device__ __forceinline__ v8f zero8() {
  v8f z;
#pragma unroll
  for (int i = 0; i < 8; ++i) z[i] = 0.f;
  return z;
}

// 16-bit A/B fragment (16x32): element e<8 -> K = g8+e, e>=8 -> K = 16+g8+(e-8)
// i.e. two contiguous 8-half (16B) runs -> two ds_load_b128 per fragment.
__device__ __forceinline__ v16h load_frag(const _Float16* row, int g8) {
  union { v16h v; v8h h[2]; } f;
  f.h[0] = *(const v8h*)(row + g8);
  f.h[1] = *(const v8h*)(row + 16 + g8);
  return f.v;
}

// ---------------------------------------------------------------------------
// Tiled GEMM:  C[M,N] = A[M,K] * B[K,N] (+bias), fp32 in/out, f16 WMMA compute
// 128x64 tile / 256 threads (8 waves, one 16x64 strip each).
// B staged TRANSPOSED in LDS; all fragments preloaded so the 4 WMMAs issue
// back-to-back after a single LDS wait.
// ---------------------------------------------------------------------------
__global__ __launch_bounds__(256)
void gemm_wmma(const float* __restrict__ A, const float* __restrict__ B,
               const float* __restrict__ bias, float* __restrict__ C,
               int M, int N, int K) {
  __shared__ __align__(16) _Float16 As[128][32];  // [m][k]
  __shared__ __align__(16) _Float16 Bt[64][32];   // [n][k] (transposed)

  const int col0 = blockIdx.x * 64;
  const int row0 = blockIdx.y * 128;
  const int tid  = threadIdx.x;
  const int wave = tid >> 5;            // 0..7 -> 16-row strip each
  const int lane = tid & 31;
  const int ln   = lane & 15;
  const int g8   = (lane >> 4) << 3;

  v8f acc[4];
#pragma unroll
  for (int nt = 0; nt < 4; ++nt) acc[nt] = zero8();

  for (int k0 = 0; k0 < K; k0 += 32) {
    // ---- stage A (128x32), fp32 -> f16 ----
    if (row0 + 128 <= M && k0 + 32 <= K) {
      for (int t = tid; t < 1024; t += 256) {
        int r = t >> 3, c4 = (t & 7) << 2;
        const float4 q = *(const float4*)(A + (size_t)(row0 + r) * K + k0 + c4);
        v4h h;
        h[0] = (_Float16)q.x; h[1] = (_Float16)q.y;
        h[2] = (_Float16)q.z; h[3] = (_Float16)q.w;
        *(v4h*)&As[r][c4] = h;
      }
    } else {
      for (int t = tid; t < 4096; t += 256) {
        int r = t >> 5, c = t & 31;
        int gr = row0 + r, gc = k0 + c;
        As[r][c] = (_Float16)((gr < M && gc < K) ? A[(size_t)gr * K + gc] : 0.f);
      }
    }
    // ---- stage B transposed (Bt[n][k]); global reads stay coalesced over n ----
    if (col0 + 64 <= N && k0 + 32 <= K) {
      for (int t = tid; t < 512; t += 256) {
        int c = t & 63, r4 = (t >> 6) << 2;
        v4h h;
#pragma unroll
        for (int i = 0; i < 4; ++i)
          h[i] = (_Float16)B[(size_t)(k0 + r4 + i) * N + col0 + c];
        *(v4h*)&Bt[c][r4] = h;
      }
    } else {
      for (int t = tid; t < 2048; t += 256) {
        int c = t & 63, r = t >> 6;
        int gr = k0 + r, gc = col0 + c;
        Bt[c][r] = (_Float16)((gr < K && gc < N) ? B[(size_t)gr * N + gc] : 0.f);
      }
    }
    // prefetch next K-tile while the WMMAs run on this one
    if (k0 + 32 < K) {
      __builtin_prefetch(A + (size_t)(row0 + (tid >> 1)) * K + k0 + 32, 0, 1);
      __builtin_prefetch(B + (size_t)(k0 + 32 + (tid >> 3)) * N + col0, 0, 1);
    }
    __syncthreads();

    // preload all fragments, then issue WMMAs back-to-back
    v16h bfr[4];
#pragma unroll
    for (int nt = 0; nt < 4; ++nt)
      bfr[nt] = load_frag(&Bt[(nt << 4) + ln][0], g8);
    v16h af = load_frag(&As[(wave << 4) + ln][0], g8);
#pragma unroll
    for (int nt = 0; nt < 4; ++nt)
      acc[nt] = __builtin_amdgcn_wmma_f32_16x16x32_f16(
          false, af, false, bfr[nt], (short)0, acc[nt], false, false);
    __syncthreads();
  }

  // C/D layout: lane ln = column, rows r + 8*(lane/16)
#pragma unroll
  for (int nt = 0; nt < 4; ++nt) {
    int gc = col0 + (nt << 4) + ln;
    if (gc >= N) continue;
    float bv = bias ? bias[gc] : 0.f;
#pragma unroll
    for (int r = 0; r < 8; ++r) {
      int gr = row0 + (wave << 4) + g8 + r;
      if (gr < M) C[(size_t)gr * N + gc] = acc[nt][r] + bv;
    }
  }
}

// ---------------------------------------------------------------------------
// Flash attention with WMMA. Block = (64-query tile, head), 4 waves.
// V staged TRANSPOSED (Vt[d][key]); B fragments preloaded per K-chunk so the
// WMMA chain issues back-to-back.
// ---------------------------------------------------------------------------
__global__ __launch_bounds__(128)
void attn_wmma(const float* __restrict__ qkv, float* __restrict__ o,
               int windowed) {
  __shared__ __align__(16) _Float16 Qs[64][kADP];
  __shared__ __align__(16) _Float16 Ks[64][kADP];
  __shared__ __align__(16) _Float16 Vt[kAD][64];   // [dim][key]
  __shared__ __align__(16) _Float16 Ps[4][16][64]; // per-wave P strip

  const int qt   = blockIdx.x;
  const int h    = blockIdx.y;
  const int tid  = threadIdx.x;
  const int wave = tid >> 5;
  const int lane = tid & 31;
  const int ln   = lane & 15;
  const int g8   = (lane >> 4) << 3;
  const float scale = 0.111803398875f;   // 1/sqrt(80), folded into Q

  // stage Q (scaled) + zero pad dims 80..95
  for (int t = tid; t < 64 * 20; t += 128) {
    int r = t / 20, c4 = (t % 20) << 2;
    int s = qt * 64 + r;
    const float4 q = *(const float4*)(qkv + ((size_t)(s * 3 + 0) * kNH + h) * kAD + c4);
    v4h hv;
    hv[0] = (_Float16)(q.x * scale); hv[1] = (_Float16)(q.y * scale);
    hv[2] = (_Float16)(q.z * scale); hv[3] = (_Float16)(q.w * scale);
    *(v4h*)&Qs[r][c4] = hv;
  }
  {
    v4h zh; zh[0] = (_Float16)0.f; zh[1] = (_Float16)0.f;
    zh[2] = (_Float16)0.f; zh[3] = (_Float16)0.f;
    for (int t = tid; t < 64 * 4; t += 128) {
      int r = t >> 2, c4 = 80 + ((t & 3) << 2);
      *(v4h*)&Qs[r][c4] = zh;
      *(v4h*)&Ks[r][c4] = zh;     // K pad region is loop-invariant too
    }
  }

  v8f oacc[5];
#pragma unroll
  for (int d = 0; d < 5; ++d) oacc[d] = zero8();
  float m[8], l[8];
#pragma unroll
  for (int r = 0; r < 8; ++r) { m[r] = -3.0e38f; l[r] = 0.f; }

  const int kt0 = windowed ? qt : 0;
  const int kt1 = windowed ? qt + 1 : kNW;

  for (int kt = kt0; kt < kt1; ++kt) {
    for (int t = tid; t < 64 * 20; t += 128) {
      int r = t / 20, c4 = (t % 20) << 2;
      int s = kt * 64 + r;
      const float4 q = *(const float4*)(qkv + ((size_t)(s * 3 + 1) * kNH + h) * kAD + c4);
      v4h hv;
      hv[0] = (_Float16)q.x; hv[1] = (_Float16)q.y;
      hv[2] = (_Float16)q.z; hv[3] = (_Float16)q.w;
      *(v4h*)&Ks[r][c4] = hv;
    }
    for (int t = tid; t < 64 * 20; t += 128) {
      int r = t / 20, c4 = (t % 20) << 2;
      int s = kt * 64 + r;
      const float4 q = *(const float4*)(qkv + ((size_t)(s * 3 + 2) * kNH + h) * kAD + c4);
      Vt[c4 + 0][r] = (_Float16)q.x;
      Vt[c4 + 1][r] = (_Float16)q.y;
      Vt[c4 + 2][r] = (_Float16)q.z;
      Vt[c4 + 3][r] = (_Float16)q.w;
    }
    __syncthreads();

    // S = (Q*scale) . K^T  — B[d][n] = K[n][d] is row-contiguous in Ks
    v8f sacc[4];
#pragma unroll
    for (int nt = 0; nt < 4; ++nt) sacc[nt] = zero8();
#pragma unroll
    for (int kc = 0; kc < 3; ++kc) {
      v16h bfr[4];
#pragma unroll
      for (int nt = 0; nt < 4; ++nt)
        bfr[nt] = load_frag(&Ks[(nt << 4) + ln][kc * 32], g8);
      v16h af = load_frag(&Qs[(wave << 4) + ln][kc * 32], g8);
#pragma unroll
      for (int nt = 0; nt < 4; ++nt)
        sacc[nt] = __builtin_amdgcn_wmma_f32_16x16x32_f16(
            false, af, false, bfr[nt], (short)0, sacc[nt], false, false);
    }

    // online softmax (rows in accum VGPRs, columns across 16 lanes per half)
#pragma unroll
    for (int r = 0; r < 8; ++r) {
      float v = sacc[0][r];
      v = fmaxf(v, sacc[1][r]);
      v = fmaxf(v, sacc[2][r]);
      v = fmaxf(v, sacc[3][r]);
      v = fmaxf(v, __shfl_xor(v, 1, 32));
      v = fmaxf(v, __shfl_xor(v, 2, 32));
      v = fmaxf(v, __shfl_xor(v, 4, 32));
      v = fmaxf(v, __shfl_xor(v, 8, 32));
      float mn = fmaxf(m[r], v);
      float alpha = __expf(m[r] - mn);
      m[r] = mn;
      l[r] *= alpha;
#pragma unroll
      for (int d = 0; d < 5; ++d) oacc[d][r] *= alpha;
    }
#pragma unroll
    for (int nt = 0; nt < 4; ++nt) {
#pragma unroll
      for (int r = 0; r < 8; ++r)
        sacc[nt][r] = __expf(sacc[nt][r] - m[r]);
    }
#pragma unroll
    for (int r = 0; r < 8; ++r) {
      float sum = sacc[0][r] + sacc[1][r] + sacc[2][r] + sacc[3][r];
      sum += __shfl_xor(sum, 1, 32);
      sum += __shfl_xor(sum, 2, 32);
      sum += __shfl_xor(sum, 4, 32);
      sum += __shfl_xor(sum, 8, 32);
      l[r] += sum;
    }

    // spill P (C/D layout) to per-wave LDS strip; reload in A-fragment order
#pragma unroll
    for (int nt = 0; nt < 4; ++nt) {
#pragma unroll
      for (int r = 0; r < 8; ++r)
        Ps[wave][g8 + r][(nt << 4) + ln] = (_Float16)sacc[nt][r];
    }
    __syncthreads();

    // O += P . V  — B[key][d] = Vt[d][key] row-contiguous
#pragma unroll
    for (int kc = 0; kc < 2; ++kc) {
      v16h bfr[5];
#pragma unroll
      for (int dt = 0; dt < 5; ++dt)
        bfr[dt] = load_frag(&Vt[(dt << 4) + ln][kc * 32], g8);
      v16h af = load_frag(&Ps[wave][ln][kc * 32], g8);
#pragma unroll
      for (int dt = 0; dt < 5; ++dt)
        oacc[dt] = __builtin_amdgcn_wmma_f32_16x16x32_f16(
            false, af, false, bfr[dt], (short)0, oacc[dt], false, false);
    }
    __syncthreads();
  }

#pragma unroll
  for (int dt = 0; dt < 5; ++dt) {
    int col = (dt << 4) + ln;
#pragma unroll
    for (int r = 0; r < 8; ++r) {
      int row = qt * 64 + (wave << 4) + g8 + r;
      o[(size_t)row * kHID + h * kAD + col] = oacc[dt][r] / l[r];
    }
  }
}

// ---------------------------------------------------------------------------
// Elementwise / data-movement kernels (pure bandwidth)
// ---------------------------------------------------------------------------
__global__ void gather_rows_k(const float* __restrict__ in, float* __restrict__ out,
                              const int* __restrict__ idx, int mu, int w, int rows) {
  long total = (long)rows * w;
  for (long i = (long)blockIdx.x * blockDim.x + threadIdx.x; i < total;
       i += (long)gridDim.x * blockDim.x) {
    int s = (int)(i / w), c = (int)(i % w);
    int src = idx[s / mu] * mu + (s % mu);
    out[i] = in[(size_t)src * w + c];
  }
}

__global__ void rope_make_k(const float* __restrict__ rpe,
                            float* __restrict__ cosb, float* __restrict__ sinb) {
  int s = blockIdx.x, j = threadIdx.x;              // 80 threads
  float a = rpe[s * kROT + (j % kROT)];
  cosb[s * 2 * kROT + j] = cosf(a);
  sinb[s * 2 * kROT + j] = sinf(a);
}

__global__ void rope_apply_k(float* __restrict__ qkv,
                             const float* __restrict__ cosb,
                             const float* __restrict__ sinb) {
  int s = blockIdx.x;
  for (int t = threadIdx.x; t < kNH * kROT; t += blockDim.x) {
    int h = t / kROT, d = t % kROT;
    float c  = cosb[s * 2 * kROT + d];   // emb repeats: cos[d] == cos[d+40]
    float sn = sinb[s * 2 * kROT + d];
#pragma unroll
    for (int which = 0; which < 2; ++which) {        // q then k
      float* p = qkv + ((size_t)(s * 3 + which) * kNH + h) * kAD + d;
      float a = p[0], b = p[kROT];
      p[0]    = a * c - b * sn;
      p[kROT] = b * c + a * sn;
    }
  }
}

__global__ __launch_bounds__(256)
void rmsnorm_k(const float* __restrict__ x, const float* __restrict__ w,
               float* __restrict__ y, int d) {
  int row = blockIdx.x;
  __shared__ float red[8];
  __shared__ float rstd;
  float s = 0.f;
  for (int c = threadIdx.x; c < d; c += 256) {
    float v = x[(size_t)row * d + c];
    s += v * v;
  }
#pragma unroll
  for (int msk = 16; msk >= 1; msk >>= 1) s += __shfl_xor(s, msk, 32);
  if ((threadIdx.x & 31) == 0) red[threadIdx.x >> 5] = s;
  __syncthreads();
  if (threadIdx.x == 0) {
    float t = 0.f;
#pragma unroll
    for (int i = 0; i < 8; ++i) t += red[i];
    rstd = rsqrtf(t / (float)d + 1e-6f);
  }
  __syncthreads();
  float r = rstd;
  for (int c = threadIdx.x; c < d; c += 256)
    y[(size_t)row * d + c] = x[(size_t)row * d + c] * r * w[c];
}

__global__ void add_k(float* __restrict__ y, const float* __restrict__ x, long n) {
  for (long i = (long)blockIdx.x * blockDim.x + threadIdx.x; i < n;
       i += (long)gridDim.x * blockDim.x)
    y[i] += x[i];
}

__global__ void silu_mul_k(float* __restrict__ g, const float* __restrict__ u, long n) {
  for (long i = (long)blockIdx.x * blockDim.x + threadIdx.x; i < n;
       i += (long)gridDim.x * blockDim.x) {
    float v = g[i];
    g[i] = (v / (1.f + __expf(-v))) * u[i];
  }
}

__global__ void gelu_k(float* __restrict__ x, long n) {
  for (long i = (long)blockIdx.x * blockDim.x + threadIdx.x; i < n;
       i += (long)gridDim.x * blockDim.x) {
    float v = x[i];
    x[i] = 0.5f * v * (1.f + erff(v * 0.70710678118654752f));
  }
}

// ---------------------------------------------------------------------------
extern "C" void kernel_launch(void* const* d_in, const int* in_sizes, int n_in,
                              void* d_out, int out_size, void* d_ws, size_t ws_size,
                              hipStream_t stream) {
  (void)in_sizes; (void)n_in; (void)out_size; (void)ws_size;

  const float* hidden  = (const float*)d_in[0];
  const float* rpe     = (const float*)d_in[1];
  const int*   widx    = (const int*)d_in[5];
  const int*   rwidx   = (const int*)d_in[6];
  const float* patch_w = (const float*)d_in[7];
  const float* norm1_w = (const float*)d_in[8];
  const float* norm2_w = (const float*)d_in[9];
  const float* qkv_w   = (const float*)d_in[10];
  const float* qkv_b   = (const float*)d_in[11];
  const float* proj_w  = (const float*)d_in[12];
  const float* proj_b  = (const float*)d_in[13];
  const float* gate_w  = (const float*)d_in[14];
  const float* gate_b  = (const float*)d_in[15];
  const float* up_w    = (const float*)d_in[16];
  const float* up_b    = (const float*)d_in[17];
  const float* down_w  = (const float*)d_in[18];
  const float* down_b  = (const float*)d_in[19];
  const float* lnq_w   = (const float*)d_in[20];
  const float* mlp0_w  = (const float*)d_in[21];
  const float* mlp0_b  = (const float*)d_in[22];
  const float* mlp2_w  = (const float*)d_in[23];
  const float* mlp2_b  = (const float*)d_in[24];

  float* ws   = (float*)d_ws;
  float* tmp0 = ws;                                    // S*HID (patch pre-gather)
  float* x    = tmp0 + (size_t)kS * kHID;              // S*HID
  float* xn   = x    + (size_t)kS * kHID;              // S*HID
  float* qkv  = xn   + (size_t)kS * kHID;              // S*3*HID
  float* ao   = qkv  + (size_t)kS * 3 * kHID;          // S*HID
  float* t2   = ao   + (size_t)kS * kHID;              // S*HID
  float* gbuf = t2   + (size_t)kS * kHID;              // S*INTER
  float* ubuf = gbuf + (size_t)kS * kINTER;            // S*INTER
  float* rpeg = ubuf + (size_t)kS * kINTER;            // S*ROT
  float* cosb = rpeg + (size_t)kS * kROT;              // S*80
  float* sinb = cosb + (size_t)kS * 2 * kROT;          // S*80
  float* m0   = sinb + (size_t)kS * 2 * kROT;          // G*MERGED (== S*HID)
  float* m2   = m0   + (size_t)kS * kHID;              // G*OUT

  auto gemm = [&](const float* A, const float* B, const float* bias, float* C,
                  int M, int N, int K) {
    dim3 grid((N + 63) / 64, (M + 127) / 128);
    gemm_wmma<<<grid, 256, 0, stream>>>(A, B, bias, C, M, N, K);
  };
  auto blocks = [](long n) {
    long b = (n + 255) / 256;
    if (b > 16384) b = 16384;
    return (int)b;
  };

  // patch embed + window-index gather + rope tables
  gemm(hidden, patch_w, nullptr, tmp0, kS, kHID, kINCH);
  gather_rows_k<<<blocks((long)kS * kHID), 256, 0, stream>>>(tmp0, x, widx, kMU, kHID, kS);
  gather_rows_k<<<blocks((long)kS * kROT), 256, 0, stream>>>(rpe, rpeg, widx, kMU, kROT, kS);
  rope_make_k<<<kS, 2 * kROT, 0, stream>>>(rpeg, cosb, sinb);

  for (int i = 0; i < 4; ++i) {
    int windowed = (i == 3) ? 0 : 1;
    rmsnorm_k<<<kS, 256, 0, stream>>>(x, norm1_w + (size_t)i * kHID, xn, kHID);
    gemm(xn, qkv_w + (size_t)i * kHID * 3 * kHID, qkv_b + (size_t)i * 3 * kHID,
         qkv, kS, 3 * kHID, kHID);
    rope_apply_k<<<kS, 256, 0, stream>>>(qkv, cosb, sinb);
    attn_wmma<<<dim3(kNW, kNH), 128, 0, stream>>>(qkv, ao, windowed);
    gemm(ao, proj_w + (size_t)i * kHID * kHID, proj_b + (size_t)i * kHID,
         t2, kS, kHID, kHID);
    add_k<<<blocks((long)kS * kHID), 256, 0, stream>>>(x, t2, (long)kS * kHID);

    rmsnorm_k<<<kS, 256, 0, stream>>>(x, norm2_w + (size_t)i * kHID, xn, kHID);
    gemm(xn, gate_w + (size_t)i * kHID * kINTER, gate_b + (size_t)i * kINTER,
         gbuf, kS, kINTER, kHID);
    gemm(xn, up_w + (size_t)i * kHID * kINTER, up_b + (size_t)i * kINTER,
         ubuf, kS, kINTER, kHID);
    silu_mul_k<<<blocks((long)kS * kINTER), 256, 0, stream>>>(gbuf, ubuf, (long)kS * kINTER);
    gemm(gbuf, down_w + (size_t)i * kINTER * kHID, down_b + (size_t)i * kHID,
         t2, kS, kHID, kINTER);
    add_k<<<blocks((long)kS * kHID), 256, 0, stream>>>(x, t2, (long)kS * kHID);
  }

  // merger: rms -> (G, MERGED) gemm -> gelu -> gemm -> reverse gather
  rmsnorm_k<<<kS, 256, 0, stream>>>(x, lnq_w, xn, kHID);
  gemm(xn, mlp0_w, mlp0_b, m0, kG, kMERGED, kMERGED);
  gelu_k<<<blocks((long)kG * kMERGED), 256, 0, stream>>>(m0, (long)kG * kMERGED);
  gemm(m0, mlp2_w, mlp2_b, m2, kG, kOUT, kMERGED);
  gather_rows_k<<<blocks((long)kG * kOUT), 256, 0, stream>>>(m2, (float*)d_out,
                                                             rwidx, 1, kOUT, kG);
}
